// NeuroSATSimp_2705829397332
// MI455X (gfx1250) — matrix operand order repaired and checked by reading, not compiled, not run on hardware
//
#include <hip/hip_runtime.h>
#include <hip/hip_bf16.h>
#include <stdint.h>

#define DIM 128

// ---------------- WMMA bf16 helpers (CDNA5 / gfx1250, wave32) ----------------

typedef __attribute__((ext_vector_type(16))) __bf16 v16bf;
typedef __attribute__((ext_vector_type(8)))  float  v8f;

__device__ __forceinline__ uint16_t f2bf(float x) {
  uint32_t u = __float_as_uint(x);
  u += 0x7FFFu + ((u >> 16) & 1u);   // round-to-nearest-even
  return (uint16_t)(u >> 16);
}

__device__ __forceinline__ uint64_t pack4bf(float a, float b, float c, float d) {
  return (uint64_t)f2bf(a) | ((uint64_t)f2bf(b) << 16) |
         ((uint64_t)f2bf(c) << 32) | ((uint64_t)f2bf(d) << 48);
}

union Frag { v16bf v; uint32_t u[8]; };

// A-matrix 16x32 bf16 fragment from row-major bf16 LDS tile [128][128].
// ISA layout: lane = 16h + m; VGPR p holds K = 2*(p&3) + 16*(p>>2) + 8h (+0,+1).
__device__ __forceinline__ v16bf load_a_frag(const uint16_t* act, int row0,
                                             int kc, int m, int h) {
  Frag f;
  const uint32_t* b32 = (const uint32_t*)act;
  int row = row0 + m;
#pragma unroll
  for (int p = 0; p < 8; ++p) {
    int K = 32 * kc + 2 * (p & 3) + 16 * (p >> 2) + 8 * h;
    f.u[p] = b32[(row * DIM + K) >> 1];
  }
  return f.v;
}

// B-matrix 32x16 bf16 fragment for output column `col` from LDS weights stored
// TRANSPOSED [n][k].  ISA layout: lane = 16h + n; VGPR p holds K = 2*(p&3) + 16*(p>>2) + 8h.
__device__ __forceinline__ v16bf load_b_frag(const uint16_t* wt, int kc,
                                             int col, int h) {
  Frag f;
  const uint32_t* b32 = (const uint32_t*)wt;
#pragma unroll
  for (int p = 0; p < 8; ++p) {
    int K = 32 * kc + 2 * (p & 3) + 16 * (p >> 2) + 8 * h;
    f.u[p] = b32[(col * DIM + K) >> 1];
  }
  return f.v;
}

__device__ __forceinline__ v8f wmma_bf16(v16bf a, v16bf b, v8f c) {
  return __builtin_amdgcn_wmma_f32_16x16x32_bf16(false, a, false, b,
                                                 (short)0, c, false, false);
}

// ---------------- CDNA5 async LDS<->global DMA (ASYNCcnt path) ----------------
// Dynamic LDS base is offset 0 in these kernels (no static LDS declared).

__device__ __forceinline__ void ldsa_load_b128(uint32_t lds_byte_off,
                                               const void* gsrc) {
  asm volatile("global_load_async_to_lds_b128 %0, %1, off"
               :: "v"(lds_byte_off), "v"(gsrc) : "memory");
}

__device__ __forceinline__ void ldsa_store_b128(void* gdst,
                                                uint32_t lds_byte_off) {
  asm volatile("global_store_async_from_lds_b128 %0, %1, off"
               :: "v"(gdst), "v"(lds_byte_off) : "memory");
}

__device__ __forceinline__ void wait_async0() {
  asm volatile("s_wait_asynccnt 0x0" ::: "memory");
}

__device__ __forceinline__ float inorm1(float x, float su, float sq,
                                        float w, float b, float inv) {
  float m = su * inv;
  float var = sq * inv - m * m;
  return w * (x - m) * rsqrtf(var + 1e-6f) + b;
}

// ---------------- small utility kernels ----------------

__global__ void k_zero(float* __restrict__ p, size_t n) {
  size_t i = (size_t)blockIdx.x * blockDim.x + threadIdx.x;
  size_t s = (size_t)gridDim.x * blockDim.x;
  for (; i < n; i += s) p[i] = 0.f;
}

// f32 [k][n] -> bf16 transposed [n][k] (done ONCE per weight matrix)
__global__ void k_cvtw(const float* __restrict__ W, uint16_t* __restrict__ O) {
  int idx = blockIdx.x * blockDim.x + threadIdx.x;
  if (idx < DIM * DIM) {
    int k = idx >> 7, n = idx & 127;
    O[n * DIM + k] = f2bf(W[idx]);
  }
}

__global__ void k_deg(const int* __restrict__ cl, const int* __restrict__ cc,
                      float* __restrict__ degl, float* __restrict__ degc,
                      int n_cells) {
  int i = blockIdx.x * blockDim.x + threadIdx.x;
  int s = gridDim.x * blockDim.x;
  for (; i < n_cells; i += s) {
    atomicAdd(&degl[cl[i]], 1.f);
    atomicAdd(&degc[cc[i]], 1.f);
  }
}

__global__ void k_init(const float* __restrict__ deg, const float* __restrict__ Wv,
                       const float* __restrict__ bv, float* __restrict__ out,
                       int n_rows) {
  size_t total = (size_t)n_rows * DIM;
  size_t stride = (size_t)gridDim.x * blockDim.x;
  for (size_t idx = (size_t)blockIdx.x * blockDim.x + threadIdx.x; idx < total;
       idx += stride) {
    int r = (int)(idx >> 7), c = (int)(idx & 127);
    out[idx] = deg[r] * Wv[c] + bv[c];
  }
}

// ---------------- single GEMM: Y = X @ W + b (WMMA bf16, f32 accum) ----------

__global__ __launch_bounds__(256) void k_gemm_rows(
    const float* __restrict__ X, const uint16_t* __restrict__ Wbf,
    const float* __restrict__ B, float* __restrict__ Y, int n_rows) {
  extern __shared__ uint16_t smem[];
  uint16_t* s_w   = smem;             // 128x128 bf16, transposed [n][k]
  uint16_t* s_act = smem + DIM * DIM; // 128x128 bf16, row-major
  int tid = threadIdx.x;
  int row0 = blockIdx.x * 128;
  bool full = (row0 + 128 <= n_rows);

  // weights: contiguous bf16 copy via async LDS DMA (overlaps act staging)
  for (int q = tid; q < (DIM * DIM) / 8; q += 256)
    ldsa_load_b128((uint32_t)(q * 16), Wbf + (size_t)q * 8);

  // activations: float4 loads, bf16 convert, 8B LDS stores
  for (int q = tid; q < (DIM * DIM) / 4; q += 256) {
    int r = q >> 5, c4 = (q & 31) << 2;
    int g = row0 + r;
    float x0 = 0.f, x1 = 0.f, x2 = 0.f, x3 = 0.f;
    if (full || g < n_rows) {
      float4 v = *(const float4*)(X + (size_t)g * DIM + c4);
      x0 = v.x; x1 = v.y; x2 = v.z; x3 = v.w;
    }
    *(uint64_t*)(s_act + r * DIM + c4) = pack4bf(x0, x1, x2, x3);
  }
  wait_async0();
  __syncthreads();

  int lane = tid & 31, h = lane >> 4, n = lane & 15;
  int rbase = (tid >> 5) * 16;   // 8 waves, 16-row stripe each

  v16bf a[4];
#pragma unroll
  for (int kc = 0; kc < 4; ++kc) a[kc] = load_a_frag(s_act, rbase, kc, n, h);

  v8f acc[8];
#pragma unroll
  for (int nt = 0; nt < 8; ++nt) {
    v8f c = {0.f, 0.f, 0.f, 0.f, 0.f, 0.f, 0.f, 0.f};
#pragma unroll
    for (int kc = 0; kc < 4; ++kc)
      c = wmma_bf16(a[kc], load_b_frag(s_w, kc, nt * 16 + n, h), c);
    acc[nt] = c;
  }

  __syncthreads();                       // everyone done with s_w/s_act
  if (full) {
    float* s_out = (float*)smem;         // 64 KB f32 tile, reuse whole LDS
#pragma unroll
    for (int nt = 0; nt < 8; ++nt) {
      int col = nt * 16 + n;
      float bias = B[col];
#pragma unroll
      for (int r = 0; r < 8; ++r)
        s_out[(rbase + r + 8 * h) * DIM + col] = acc[nt][r] + bias;
    }
    __syncthreads();
    for (int q = tid; q < (DIM * DIM) / 4; q += 256)
      ldsa_store_b128(Y + (size_t)row0 * DIM + q * 4, (uint32_t)(q * 16));
    wait_async0();
  } else {
#pragma unroll
    for (int nt = 0; nt < 8; ++nt) {
      int col = nt * 16 + n;
      float bias = B[col];
#pragma unroll
      for (int r = 0; r < 8; ++r) {
        int grow = row0 + rbase + r + 8 * h;
        if (grow < n_rows) Y[(size_t)grow * DIM + col] = acc[nt][r] + bias;
      }
    }
  }
}

// ------- fused [optional inorm] -> Lin+ReLU -> Lin+ReLU -> Lin [+res] -------

__global__ __launch_bounds__(256) void k_mlp3(
    const float* __restrict__ X,
    const float* __restrict__ ssum, const float* __restrict__ ssq,
    const float* __restrict__ nw, const float* __restrict__ nb,
    int rows_per_prob, float inv_cnt,
    const uint16_t* __restrict__ Wbf,   // 3 contiguous transposed bf16 matrices
    const float* __restrict__ B1, const float* __restrict__ B2,
    const float* __restrict__ B3,
    const float* __restrict__ RES, float* __restrict__ Y, int n_rows) {
  extern __shared__ uint16_t smem[];
  uint16_t* s_w   = smem;                  // 3 x 128x128 bf16, transposed
  uint16_t* s_act = smem + 3 * DIM * DIM;  // 128x128 bf16 row-major
  int tid = threadIdx.x;
  int row0 = blockIdx.x * 128;
  bool full = (row0 + 128 <= n_rows);

  // all three layer weights via async LDS DMA
  for (int q = tid; q < (3 * DIM * DIM) / 8; q += 256)
    ldsa_load_b128((uint32_t)(q * 16), Wbf + (size_t)q * 8);

  bool do_norm = (ssum != nullptr);
  for (int q = tid; q < (DIM * DIM) / 4; q += 256) {
    int r = q >> 5, c4 = (q & 31) << 2;
    int g = row0 + r;
    float x0 = 0.f, x1 = 0.f, x2 = 0.f, x3 = 0.f;
    if (full || g < n_rows) {
      float4 v = *(const float4*)(X + (size_t)g * DIM + c4);
      x0 = v.x; x1 = v.y; x2 = v.z; x3 = v.w;
      if (do_norm) {
        int p = g / rows_per_prob;
        const float* su = ssum + p * DIM + c4;
        const float* sq = ssq + p * DIM + c4;
        x0 = inorm1(x0, su[0], sq[0], nw[c4 + 0], nb[c4 + 0], inv_cnt);
        x1 = inorm1(x1, su[1], sq[1], nw[c4 + 1], nb[c4 + 1], inv_cnt);
        x2 = inorm1(x2, su[2], sq[2], nw[c4 + 2], nb[c4 + 2], inv_cnt);
        x3 = inorm1(x3, su[3], sq[3], nw[c4 + 3], nb[c4 + 3], inv_cnt);
      }
    }
    *(uint64_t*)(s_act + r * DIM + c4) = pack4bf(x0, x1, x2, x3);
  }
  wait_async0();
  __syncthreads();

  int lane = tid & 31, h = lane >> 4, n = lane & 15;
  int rbase = (tid >> 5) * 16;
  const float* Bs[3] = {B1, B2, B3};

  for (int layer = 0; layer < 3; ++layer) {
    v16bf a[4];
#pragma unroll
    for (int kc = 0; kc < 4; ++kc) a[kc] = load_a_frag(s_act, rbase, kc, n, h);
    v8f acc[8];
#pragma unroll
    for (int nt = 0; nt < 8; ++nt) {
      v8f c = {0.f, 0.f, 0.f, 0.f, 0.f, 0.f, 0.f, 0.f};
#pragma unroll
      for (int kc = 0; kc < 4; ++kc)
        c = wmma_bf16(a[kc],
                      load_b_frag(s_w + layer * DIM * DIM, kc, nt * 16 + n, h),
                      c);
      acc[nt] = c;
    }
    if (layer < 2) {
      __syncthreads();
#pragma unroll
      for (int nt = 0; nt < 8; ++nt) {
        float bias = Bs[layer][nt * 16 + n];
#pragma unroll
        for (int r = 0; r < 8; ++r) {
          float v = fmaxf(acc[nt][r] + bias, 0.f);
          s_act[(rbase + r + 8 * h) * DIM + nt * 16 + n] = f2bf(v);
        }
      }
      __syncthreads();
    } else {
      __syncthreads();                 // all waves done reading s_w
      if (full) {
        float* s_out = (float*)smem;   // 64 KB f32 tile reusing weight LDS
#pragma unroll
        for (int nt = 0; nt < 8; ++nt) {
          int col = nt * 16 + n;
          float bias = B3[col];
#pragma unroll
          for (int r = 0; r < 8; ++r) {
            int lrow = rbase + r + 8 * h;
            float v = acc[nt][r] + bias;
            if (RES) v += RES[(size_t)(row0 + lrow) * DIM + col];
            s_out[lrow * DIM + col] = v;
          }
        }
        __syncthreads();
        for (int q = tid; q < (DIM * DIM) / 4; q += 256)
          ldsa_store_b128(Y + (size_t)row0 * DIM + q * 4, (uint32_t)(q * 16));
        wait_async0();
      } else {
#pragma unroll
        for (int nt = 0; nt < 8; ++nt) {
          int col = nt * 16 + n;
          float bias = B3[col];
#pragma unroll
          for (int r = 0; r < 8; ++r) {
            int grow = row0 + rbase + r + 8 * h;
            if (grow < n_rows) {
              float v = acc[nt][r] + bias;
              if (RES) v += RES[(size_t)grow * DIM + col];
              Y[(size_t)grow * DIM + col] = v;
            }
          }
        }
      }
    }
  }
}

// ---- LC[c] = sum of Lp over the clause's 3 literals; accumulate inorm stats --

__global__ __launch_bounds__(256) void k_gather3(
    const float* __restrict__ Lp, const int* __restrict__ cell_lit,
    float* __restrict__ LC, float* __restrict__ ssum, float* __restrict__ ssq,
    int rows_per_prob, int n_clauses) {
  __shared__ float rs[2][DIM], rq[2][DIM];
  int tid = threadIdx.x;
  int ch = tid & 127, sub = tid >> 7;
  int c0 = blockIdx.x * 32;
  float ls = 0.f, lq = 0.f;
  for (int j = sub; j < 32; j += 2) {
    int c = c0 + j;
    if (c < n_clauses) {
      int cn = c + 2;                       // prefetch next clause's rows
      if (cn < n_clauses) {
        __builtin_prefetch(&Lp[(size_t)cell_lit[3 * cn] * DIM + ch], 0, 1);
        __builtin_prefetch(&Lp[(size_t)cell_lit[3 * cn + 1] * DIM + ch], 0, 1);
        __builtin_prefetch(&Lp[(size_t)cell_lit[3 * cn + 2] * DIM + ch], 0, 1);
      }
      int l0 = cell_lit[3 * c], l1 = cell_lit[3 * c + 1], l2 = cell_lit[3 * c + 2];
      float v = Lp[(size_t)l0 * DIM + ch] + Lp[(size_t)l1 * DIM + ch] +
                Lp[(size_t)l2 * DIM + ch];
      LC[(size_t)c * DIM + ch] = v;
      ls += v; lq += v * v;
    }
  }
  rs[sub][ch] = ls; rq[sub][ch] = lq;
  __syncthreads();
  if (sub == 0) {
    int p = c0 / rows_per_prob;
    atomicAdd(&ssum[p * DIM + ch], rs[0][ch] + rs[1][ch]);
    atomicAdd(&ssq[p * DIM + ch], rq[0][ch] + rq[1][ch]);
  }
}

// ---- CL[lit] += Cm[clause] over cells (scatter via L2 atomics) ----

__global__ __launch_bounds__(128) void k_scatter(
    const float* __restrict__ Cm, const int* __restrict__ cell_lit,
    const int* __restrict__ cell_clause, float* __restrict__ CL, int n_cells) {
  int ch = threadIdx.x;
  for (int c = blockIdx.x; c < n_cells; c += gridDim.x) {
    int cn = c + gridDim.x;
    if (cn < n_cells)
      __builtin_prefetch(&Cm[(size_t)cell_clause[cn] * DIM + ch], 0, 1);
    int lit = cell_lit[c], cl = cell_clause[c];
    atomicAdd(&CL[(size_t)lit * DIM + ch], Cm[(size_t)cl * DIM + ch]);
  }
}

// ---- Lpre = CL + flip(L); accumulate inorm stats ----

__global__ __launch_bounds__(256) void k_lpre(
    const float* __restrict__ CL, const float* __restrict__ L,
    float* __restrict__ Lpre, float* __restrict__ ssum, float* __restrict__ ssq,
    int n_vars, int rows_per_prob, int n_lits) {
  __shared__ float rs[2][DIM], rq[2][DIM];
  int tid = threadIdx.x;
  int ch = tid & 127, sub = tid >> 7;
  int l0 = blockIdx.x * 32;
  float ls = 0.f, lq = 0.f;
  for (int j = sub; j < 32; j += 2) {
    int l = l0 + j;
    if (l < n_lits) {
      int fl = (l < n_vars) ? l + n_vars : l - n_vars;
      float v = CL[(size_t)l * DIM + ch] + L[(size_t)fl * DIM + ch];
      Lpre[(size_t)l * DIM + ch] = v;
      ls += v; lq += v * v;
    }
  }
  rs[sub][ch] = ls; rq[sub][ch] = lq;
  __syncthreads();
  if (sub == 0) {
    int p = l0 / rows_per_prob;
    atomicAdd(&ssum[p * DIM + ch], rs[0][ch] + rs[1][ch]);
    atomicAdd(&ssq[p * DIM + ch], rq[0][ch] + rq[1][ch]);
  }
}

// ---- rep[p] = mean over the problem's literals of L ----

__global__ __launch_bounds__(256) void k_rep(
    const float* __restrict__ L, float* __restrict__ rep, int rows_per_prob,
    float inv_cnt, int n_lits) {
  __shared__ float rs[2][DIM];
  int tid = threadIdx.x;
  int ch = tid & 127, sub = tid >> 7;
  int l0 = blockIdx.x * 32;
  float ls = 0.f;
  for (int j = sub; j < 32; j += 2) {
    int l = l0 + j;
    if (l < n_lits) ls += L[(size_t)l * DIM + ch];
  }
  rs[sub][ch] = ls;
  __syncthreads();
  if (sub == 0) {
    int p = l0 / rows_per_prob;
    atomicAdd(&rep[p * DIM + ch], (rs[0][ch] + rs[1][ch]) * inv_cnt);
  }
}

// ---------------- host orchestration ----------------

extern "C" void kernel_launch(void* const* d_in, const int* in_sizes, int n_in,
                              void* d_out, int out_size, void* d_ws, size_t ws_size,
                              hipStream_t stream) {
  const int* cell_lit    = (const int*)d_in[0];
  const int* cell_clause = (const int*)d_in[1];
  int n_cells   = in_sizes[0];
  int n_lits    = in_sizes[2];
  int n_clauses = in_sizes[3];
  int n_probs   = out_size / DIM;          // vote output is (n_probs, DIM)
  int n_vars    = n_lits / 2;
  int clauses_per = n_clauses / n_probs;
  int lits_per    = n_lits / n_probs;

  const float* L_init_W = (const float*)d_in[5];
  const float* L_init_b = (const float*)d_in[6];
  const float* C_init_W = (const float*)d_in[7];
  const float* C_init_b = (const float*)d_in[8];
  const float* Lproj_W  = (const float*)d_in[9];
  const float* Lproj_b  = (const float*)d_in[10];
  const float* Lmsg_W1 = (const float*)d_in[11]; const float* Lmsg_b1 = (const float*)d_in[12];
  const float* Lmsg_W2 = (const float*)d_in[13]; const float* Lmsg_b2 = (const float*)d_in[14];
  const float* Lmsg_W3 = (const float*)d_in[15]; const float* Lmsg_b3 = (const float*)d_in[16];
  const float* Cmsg_W1 = (const float*)d_in[17]; const float* Cmsg_b1 = (const float*)d_in[18];
  const float* Cmsg_W2 = (const float*)d_in[19]; const float* Cmsg_b2 = (const float*)d_in[20];
  const float* Cmsg_W3 = (const float*)d_in[21]; const float* Cmsg_b3 = (const float*)d_in[22];
  const float* Cprj_W1 = (const float*)d_in[23]; const float* Cprj_b1 = (const float*)d_in[24];
  const float* Cprj_W2 = (const float*)d_in[25]; const float* Cprj_b2 = (const float*)d_in[26];
  const float* Cprj_W3 = (const float*)d_in[27]; const float* Cprj_b3 = (const float*)d_in[28];
  const float* inorm_w1 = (const float*)d_in[29];
  const float* inorm_b1 = (const float*)d_in[30];
  const float* inorm_w2 = (const float*)d_in[31];
  const float* inorm_b2 = (const float*)d_in[32];
  const float* vote_W1 = (const float*)d_in[33]; const float* vote_b1 = (const float*)d_in[34];
  const float* vote_W2 = (const float*)d_in[35]; const float* vote_b2 = (const float*)d_in[36];
  const float* vote_W3 = (const float*)d_in[37]; const float* vote_b3 = (const float*)d_in[38];

  const int WMAT = DIM * DIM;

  // workspace layout (floats)
  float* ws = (float*)d_ws;
  size_t off = 0;
  float* Lp   = ws + off; off += (size_t)n_lits * DIM;     // Lproj out; later CL
  float* LC   = ws + off; off += (size_t)n_clauses * DIM;  // LC; later Cm; later Lpre
  float* Cst  = ws + off; off += (size_t)n_clauses * DIM;  // C state
  float* Lst  = ws + off; off += (size_t)n_lits * DIM;     // L state
  float* degl = ws + off; off += (size_t)n_lits;
  float* degc = ws + off; off += (size_t)n_clauses;
  float* ssum = ws + off; off += (size_t)n_probs * DIM;
  float* ssq  = ws + off; off += (size_t)n_probs * DIM;
  float* rep  = ws + off; off += (size_t)n_probs * DIM;
  uint16_t* wbf = (uint16_t*)(ws + off); off += (size_t)25 * WMAT / 2;
  (void)ws_size; (void)n_in;

  auto slot = [&](int s) { return wbf + (size_t)s * WMAT; };
  // bf16 weight slots: 0..3 Lproj[i]; 4+3i..6+3i Cmsg[i]; 16..18 Cproj[3];
  //                    19..21 Lmsg[3]; 22..24 vote
  for (int i = 0; i < 4; ++i) {
    k_cvtw<<<64, 256, 0, stream>>>(Lproj_W + (size_t)i * WMAT, slot(i));
    k_cvtw<<<64, 256, 0, stream>>>(Cmsg_W1 + (size_t)i * WMAT, slot(4 + 3 * i + 0));
    k_cvtw<<<64, 256, 0, stream>>>(Cmsg_W2 + (size_t)i * WMAT, slot(4 + 3 * i + 1));
    k_cvtw<<<64, 256, 0, stream>>>(Cmsg_W3 + (size_t)i * WMAT, slot(4 + 3 * i + 2));
  }
  k_cvtw<<<64, 256, 0, stream>>>(Cprj_W1 + (size_t)3 * WMAT, slot(16));
  k_cvtw<<<64, 256, 0, stream>>>(Cprj_W2 + (size_t)3 * WMAT, slot(17));
  k_cvtw<<<64, 256, 0, stream>>>(Cprj_W3 + (size_t)3 * WMAT, slot(18));
  k_cvtw<<<64, 256, 0, stream>>>(Lmsg_W1 + (size_t)3 * WMAT, slot(19));
  k_cvtw<<<64, 256, 0, stream>>>(Lmsg_W2 + (size_t)3 * WMAT, slot(20));
  k_cvtw<<<64, 256, 0, stream>>>(Lmsg_W3 + (size_t)3 * WMAT, slot(21));
  k_cvtw<<<64, 256, 0, stream>>>(vote_W1, slot(22));
  k_cvtw<<<64, 256, 0, stream>>>(vote_W2, slot(23));
  k_cvtw<<<64, 256, 0, stream>>>(vote_W3, slot(24));

  size_t gemm_smem = (size_t)2 * WMAT * sizeof(uint16_t);  // 64 KB
  size_t mlp_smem  = (size_t)4 * WMAT * sizeof(uint16_t);  // 128 KB of 320KB/WGP
  int gb_l = (n_lits + 127) / 128;
  int mb_c = (n_clauses + 127) / 128;

  // degrees + init states
  k_zero<<<1024, 256, 0, stream>>>(degl, (size_t)n_lits + n_clauses);
  k_deg<<<1024, 256, 0, stream>>>(cell_lit, cell_clause, degl, degc, n_cells);
  k_init<<<2048, 256, 0, stream>>>(degl, L_init_W, L_init_b, Lst, n_lits);
  k_init<<<2048, 256, 0, stream>>>(degc, C_init_W, C_init_b, Cst, n_clauses);

  // message-passing rounds: C <- MLP(inorm(gather(Lproj(L)))) + C
  for (int i = 0; i < 4; ++i) {
    k_gemm_rows<<<gb_l, 256, gemm_smem, stream>>>(
        Lst, slot(i), Lproj_b + (size_t)i * DIM, Lp, n_lits);
    k_zero<<<64, 256, 0, stream>>>(ssum, (size_t)2 * n_probs * DIM);
    k_gather3<<<(n_clauses + 31) / 32, 256, 0, stream>>>(
        Lp, cell_lit, LC, ssum, ssq, clauses_per, n_clauses);
    k_mlp3<<<mb_c, 256, mlp_smem, stream>>>(
        LC, ssum, ssq, inorm_w1, inorm_b1, clauses_per, 1.0f / clauses_per,
        slot(4 + 3 * i),
        Cmsg_b1 + (size_t)i * DIM, Cmsg_b2 + (size_t)i * DIM,
        Cmsg_b3 + (size_t)i * DIM,
        Cst, Cst, n_clauses);
  }

  // tail: L <- MLP(inorm(scatter(Cproj(C)) + flip(L))) + L
  const int i = 3;
  k_mlp3<<<mb_c, 256, mlp_smem, stream>>>(        // Cm = Cproj MLP(C)  -> LC buf
      Cst, nullptr, nullptr, nullptr, nullptr, 1, 1.0f,
      slot(16),
      Cprj_b1 + (size_t)i * DIM, Cprj_b2 + (size_t)i * DIM,
      Cprj_b3 + (size_t)i * DIM,
      nullptr, LC, n_clauses);
  k_zero<<<1024, 256, 0, stream>>>(Lp, (size_t)n_lits * DIM);   // CL = 0
  k_scatter<<<4096, 128, 0, stream>>>(LC, cell_lit, cell_clause, Lp, n_cells);
  k_zero<<<64, 256, 0, stream>>>(ssum, (size_t)2 * n_probs * DIM);
  k_lpre<<<(n_lits + 31) / 32, 256, 0, stream>>>(   // Lpre -> LC buf (Cm dead)
      Lp, Lst, LC, ssum, ssq, n_vars, lits_per, n_lits);
  k_mlp3<<<gb_l, 256, mlp_smem, stream>>>(
      LC, ssum, ssq, inorm_w2, inorm_b2, lits_per, 1.0f / lits_per,
      slot(19),
      Lmsg_b1 + (size_t)i * DIM, Lmsg_b2 + (size_t)i * DIM,
      Lmsg_b3 + (size_t)i * DIM,
      Lst, Lst, n_lits);

  // vote: rep = segment-mean(L); out = vote MLP(rep)
  k_zero<<<16, 256, 0, stream>>>(rep, (size_t)n_probs * DIM);
  k_rep<<<(n_lits + 31) / 32, 256, 0, stream>>>(Lst, rep, lits_per,
                                                1.0f / lits_per, n_lits);
  k_mlp3<<<(n_probs + 127) / 128, 256, mlp_smem, stream>>>(
      rep, nullptr, nullptr, nullptr, nullptr, 1, 1.0f,
      slot(22), vote_b1, vote_b2, vote_b3,
      nullptr, (float*)d_out, n_probs);
}
